// ObjectDetector_42846593745217
// MI455X (gfx1250) — matrix operand order repaired and checked by reading, not compiled, not run on hardware
//
#include <hip/hip_runtime.h>
#include <hip/hip_bf16.h>

// ---------------------------------------------------------------------------
// SSD300 forward on gfx1250 (MI455X):
//   - implicit-GEMM convs via V_WMMA_F32_16X16X32_BF16, f32 accumulate
//   - block tile 64(M) x 64(Co): 4 wave32s x 4 accumulators -> 4 wmma/step/wave
//   - A fragments: two contiguous global_load_b128 per lane (ISA A-layout)
//   - B tile: Tensor Data Mover double-buffered into LDS (s_wait_tensorcnt 1),
//     TDM pad feature (132B row pitch) kills LDS bank conflicts
//   - heads write f32 directly into d_out (NHWC flattening == permuted view)
// ---------------------------------------------------------------------------

typedef __attribute__((ext_vector_type(16))) __bf16          v16bf;
typedef __attribute__((ext_vector_type(16))) unsigned short  v16u;
typedef __attribute__((ext_vector_type(8)))  float           v8f;
typedef __attribute__((ext_vector_type(4)))  unsigned int    v4u;
typedef __attribute__((ext_vector_type(8)))  int             v8i;
typedef __attribute__((ext_vector_type(4)))  int             v4i;

__device__ __forceinline__ unsigned short f2bf(float f) {
    unsigned int u = __builtin_bit_cast(unsigned int, f);
    u += 0x7FFFu + ((u >> 16) & 1u);          // round-to-nearest-even
    return (unsigned short)(u >> 16);
}
__device__ __forceinline__ float bf2f(unsigned short u) {
    return __builtin_bit_cast(float, (unsigned int)u << 16);
}

// --------------------------- weight repack ---------------------------------
// OIHW f32 (co,ci,k,k)  ->  [(kh*k+kw)*Ci + ci][co] bf16
__global__ void repack_w_kernel(const float* __restrict__ w,
                                unsigned short* __restrict__ out,
                                int Co, int Ci, int ksz, long total) {
    long i = (long)blockIdx.x * blockDim.x + threadIdx.x;
    if (i >= total) return;
    int co = (int)(i % Co);
    long t = i / Co;
    int ci = (int)(t % Ci); t /= Ci;
    int kw = (int)(t % ksz);
    int kh = (int)(t / ksz);
    out[i] = f2bf(w[(((long)co * Ci + ci) * ksz + kh) * ksz + kw]);
}

// --------------------------- image convert ---------------------------------
__global__ void nchw2nhwc_kernel(const float* __restrict__ x,
                                 unsigned short* __restrict__ y,
                                 int N, int C, int H, int W) {
    long total = (long)N * C * H * W;
    long i = (long)blockIdx.x * blockDim.x + threadIdx.x;
    if (i >= total) return;
    int c = (int)(i % C);
    long t = i / C;
    int w = (int)(t % W); t /= W;
    int h = (int)(t % H);
    int n = (int)(t / H);
    y[i] = f2bf(x[(((long)n * C + c) * H + h) * W + w]);
}

// ----------------------------- max pool ------------------------------------
__global__ void maxpool_kernel(const unsigned short* __restrict__ x,
                               unsigned short* __restrict__ y,
                               int N, int C, int H, int W, int Ho, int Wo,
                               int k, int s, int padlo) {
    long total = (long)N * Ho * Wo * C;
    long i = (long)blockIdx.x * blockDim.x + threadIdx.x;
    if (i >= total) return;
    int c = (int)(i % C);
    long t = i / C;
    int wo = (int)(t % Wo); t /= Wo;
    int ho = (int)(t % Ho);
    int n  = (int)(t / Ho);
    int h0 = ho * s - padlo, w0 = wo * s - padlo;
    float best = -3.4e38f;
    unsigned short bb = 0;
    for (int dh = 0; dh < k; ++dh) {
        int h = h0 + dh; if (h < 0 || h >= H) continue;
        for (int dw = 0; dw < k; ++dw) {
            int w = w0 + dw; if (w < 0 || w >= W) continue;
            unsigned short u = x[(((long)n * H + h) * W + w) * C + c];
            float f = bf2f(u);
            if (f > best) { best = f; bb = u; }
        }
    }
    y[i] = bb;
}

// --------------------------- WMMA conv -------------------------------------
// Implicit GEMM: M = N*Ho*Wo, N = Co, K = k*k*Ci (32-wide steps, per tap)
// block = 128 threads = 4 waves; block tile 64(M) x 64(Co); wave: 16 x 64.
// LDS B tile rows padded by TDM: pitch 132B (66 ushort) -> bank-conflict free.
#define LB_PITCH 66
#define LB_BYTES (32 * LB_PITCH * 2)

__global__ void conv_wmma_kernel(const unsigned short* __restrict__ X,   // NHWC bf16
                                 const unsigned short* __restrict__ Wk,  // (k*k*Ci) x Co bf16
                                 const float* __restrict__ bias,
                                 unsigned short* __restrict__ Ybf,       // NHWC bf16 (or null)
                                 float* __restrict__ Yf32,               // head output (or null)
                                 int N, int H, int W, int Ci, int Co,
                                 int Ho, int Wo, int ksz, int stride, int pad, int dil,
                                 int relu, int f32_img_stride, int f32_region_off) {
    __shared__ unsigned short lB[2][32 * LB_PITCH];   // double-buffered K-major tile

    const int lane = threadIdx.x & 31;
    const int wave = threadIdx.x >> 5;
    const int co0  = blockIdx.y * 64;
    const long M   = (long)N * Ho * Wo;
    const long m0  = ((long)blockIdx.x * 4 + wave) * 16;

    // A-matrix row owned by this lane (lanes 0-15 / 16-31 both cover M=0..15)
    const long mA   = m0 + (lane & 15);
    const bool mOk  = mA < M;
    const int  nImg = mOk ? (int)(mA / ((long)Ho * Wo)) : 0;
    const int  pixA = mOk ? (int)(mA % ((long)Ho * Wo)) : 0;
    const int  hoA  = pixA / Wo, woA = pixA % Wo;
    const int  ih0  = hoA * stride - pad;
    const int  iw0  = woA * stride - pad;
    const int  hiK    = (lane >> 4) << 3;     // A: +8 K offset for lanes 16..31
    const int  khalf  = (lane >> 4) << 4;     // unused: B reads also index K with hiK
    const int  nn     = lane & 15;
    const bool ciAligned = (Ci & 31) == 0;
    const unsigned ldsAddr0 = (unsigned)(unsigned long long)(&lB[0][0]);

    // TDM: load 32(K) x 64(Co) bf16 weight tile -> LDS (K-major, 132B pitch).
    // HW zero-fills beyond tensor_dim0/1 (Co tail, K tail inside a tap).
    auto issueTdm = [&](long row0, int td1, int bufIdx) {
        unsigned long long ga = (unsigned long long)(const void*)Wk
            + (unsigned long long)(row0 * (long)Co + co0) * 2ull;
        unsigned td0 = (unsigned)(Co - co0);
        v4u g0 = { 1u, ldsAddr0 + (unsigned)bufIdx * LB_BYTES, (unsigned)ga,
                   (unsigned)((ga >> 32) & 0x01FFFFFFu) | (2u << 30) };
        v8i g1;
        // data_size=2B | pad_enable | pad_interval: every 128B add 1 dword pad
        g1[0] = (int)((1u << 16) | (1u << 20) | (4u << 22));
        g1[1] = (int)((td0 & 0xFFFFu) << 16);                                  // tensor_dim0 lo
        g1[2] = (int)(((td0 >> 16) & 0xFFFFu) | (((unsigned)td1 & 0xFFFFu) << 16));
        g1[3] = (int)((((unsigned)td1 >> 16) & 0xFFFFu) | (64u << 16));        // tile_dim0=64
        g1[4] = 32;                                                            // tile_dim1=32
        g1[5] = (int)((unsigned)Co);                                           // dim0 stride
        g1[6] = 0;
        g1[7] = 0;
        v4i g2 = { 1, 1, 0, 0 };   // tensor_dim2=1, tensor_dim3=1
        v4i g3 = { 0, 0, 0, 0 };
#if __clang_major__ >= 23
        v8i g4 = { 0, 0, 0, 0, 0, 0, 0, 0 };
        __builtin_amdgcn_tensor_load_to_lds(g0, g1, g2, g3, g4, 0);
#else
        __builtin_amdgcn_tensor_load_to_lds(g0, g1, g2, g3, 0);
#endif
    };

    const int nTotal = ksz * ksz * ((Ci + 31) >> 5);

    if (wave == 0) issueTdm(0, Ci, 0);        // prologue: stage first tile

    v8f acc[4] = {};
    int kh = 0, kw = 0, c0 = 0;
    long flatRow = 0;
    for (int it = 0; it < nTotal; ++it) {
        const int ih = ih0 + kh * dil;
        const int iw = iw0 + kw * dil;
        const bool inb = mOk && ih >= 0 && ih < H && iw >= 0 && iw < W;
        const long xbase = (((long)nImg * H + ih) * W + iw) * Ci;

        __syncthreads();                      // everyone done with buf[(it+1)&1]
        if (wave == 0) {
            if (it + 1 < nTotal) {            // prefetch next tile via TDM
                long rowN; int td1n;
                if (c0 + 32 < Ci) { rowN = flatRow + 32;        td1n = Ci - (c0 + 32); }
                else              { rowN = flatRow + (Ci - c0); td1n = Ci; }
                issueTdm(rowN, td1n, (it + 1) & 1);
                __builtin_amdgcn_s_wait_tensorcnt(1);   // current tile resident
            } else {
                __builtin_amdgcn_s_wait_tensorcnt(0);
            }
        }
        __syncthreads();                      // tile visible to all waves
        const unsigned short* lb = lB[it & 1];

        // ---- A fragment: two contiguous 16B runs per lane ----
        v16u au = {};
        if (inb) {
            if (ciAligned) {
                union { v16u v; uint4 q[2]; } U;
                U.q[0] = *(const uint4*)(X + xbase + c0 + hiK);        // K: c0+hiK..+7
                U.q[1] = *(const uint4*)(X + xbase + c0 + hiK + 16);   // K: +16..+23
                au = U.v;
                __builtin_prefetch((const void*)(X + xbase + c0 + 32), 0, 1);
            } else {                          // conv1_1 (Ci=3): guarded scalars
#pragma unroll
                for (int e = 0; e < 16; ++e) {
                    int koff = (e < 8) ? (hiK + e) : (16 + hiK + (e - 8));
                    int ci = c0 + koff;
                    unsigned short v = 0;
                    if (ci < Ci) v = X[xbase + ci];
                    au[e] = v;
                }
            }
        }
        v16bf av = __builtin_bit_cast(v16bf, au);

        // ---- 4 B fragments from LDS (conflict-free pitch), 4 WMMA ----
#pragma unroll
        for (int ch = 0; ch < 4; ++ch) {
            v16u bu = {};
#pragma unroll
            for (int e = 0; e < 16; ++e)
                bu[e] = lb[((e < 8) ? (hiK + e) : (16 + hiK + (e - 8))) * LB_PITCH + ch * 16 + nn];
            acc[ch] = __builtin_amdgcn_wmma_f32_16x16x32_bf16(
                false, av, false, __builtin_bit_cast(v16bf, bu),
                (short)0, acc[ch], false, false);
        }

        // advance (kh, kw, c0) and flat weight row
        if (c0 + 32 < Ci) { c0 += 32; flatRow += 32; }
        else {
            flatRow += Ci - c0; c0 = 0;
            if (++kw == ksz) { kw = 0; ++kh; }
        }
    }

    // bias + relu + store (C/D layout: VGPR j -> M=j (+8 hi lanes), N=lane%16)
    const int rOff = (lane >> 4) << 3;
#pragma unroll
    for (int ch = 0; ch < 4; ++ch) {
        int cc = co0 + ch * 16 + nn;
        if (cc >= Co) continue;
        float bv = bias[cc];
#pragma unroll
        for (int j = 0; j < 8; ++j) {
            long m = m0 + j + rOff;
            if (m >= M) continue;
            float v = acc[ch][j] + bv;
            if (relu) v = v > 0.0f ? v : 0.0f;
            if (Ybf) {
                Ybf[m * Co + cc] = f2bf(v);
            } else {
                long ni = m / ((long)Ho * Wo);
                long pp = m % ((long)Ho * Wo);
                Yf32[ni * f32_img_stride + f32_region_off + pp * Co + cc] = v;
            }
        }
    }
}

// ---------------------------------------------------------------------------
extern "C" void kernel_launch(void* const* d_in, const int* in_sizes, int n_in,
                              void* d_out, int out_size, void* d_ws, size_t ws_size,
                              hipStream_t stream) {
    (void)in_sizes; (void)n_in; (void)out_size; (void)ws_size;
    struct Spec { int co, ci, k; };
    static const Spec S[35] = {
        {64,3,3},{64,64,3},{128,64,3},{128,128,3},{256,128,3},{256,256,3},{256,256,3},
        {512,256,3},{512,512,3},{512,512,3},{512,512,3},{512,512,3},{512,512,3},
        {1024,512,3},{1024,1024,1},{256,1024,1},{512,256,3},{128,512,1},{256,128,3},
        {128,256,1},{256,128,3},{128,256,1},{256,128,3},
        {16,512,3},{24,1024,3},{24,512,3},{24,256,3},{16,256,3},{16,256,3},
        {84,512,3},{126,1024,3},{126,512,3},{126,256,3},{84,256,3},{84,256,3}
    };
    const int NB = 8;
    char* ws = (char*)d_ws;
    size_t off = 0;
    auto alloc = [&](size_t bytes) { size_t o = off; off += (bytes + 255) & ~(size_t)255; return o; };

    size_t o_img = alloc((size_t)NB*300*300*3*2);
    size_t o_A   = alloc((size_t)NB*300*300*64*2);
    size_t o_B   = alloc((size_t)NB*300*300*64*2);
    size_t o_f4  = alloc((size_t)NB*38*38*512*2);
    size_t o_f7  = alloc((size_t)NB*19*19*1024*2);
    size_t o_f8  = alloc((size_t)NB*10*10*512*2);
    size_t o_f9  = alloc((size_t)NB*5*5*256*2);
    size_t o_f10 = alloc((size_t)NB*3*3*256*2);
    size_t o_f11 = alloc((size_t)NB*1*1*256*2);
    size_t o_w[35];
    for (int i = 0; i < 35; ++i)
        o_w[i] = alloc((size_t)S[i].co * S[i].ci * S[i].k * S[i].k * 2);

    unsigned short* IMG = (unsigned short*)(ws + o_img);
    unsigned short* A   = (unsigned short*)(ws + o_A);
    unsigned short* Bq  = (unsigned short*)(ws + o_B);
    unsigned short* F4  = (unsigned short*)(ws + o_f4);
    unsigned short* F7  = (unsigned short*)(ws + o_f7);
    unsigned short* F8  = (unsigned short*)(ws + o_f8);
    unsigned short* F9  = (unsigned short*)(ws + o_f9);
    unsigned short* F10 = (unsigned short*)(ws + o_f10);
    unsigned short* F11 = (unsigned short*)(ws + o_f11);

    for (int i = 0; i < 35; ++i) {
        long tot = (long)S[i].co * S[i].ci * S[i].k * S[i].k;
        repack_w_kernel<<<(unsigned)((tot + 255) / 256), 256, 0, stream>>>(
            (const float*)d_in[1 + 2*i], (unsigned short*)(ws + o_w[i]),
            S[i].co, S[i].ci, S[i].k, tot);
    }
    {
        long tot = (long)NB * 3 * 300 * 300;
        nchw2nhwc_kernel<<<(unsigned)((tot + 255) / 256), 256, 0, stream>>>(
            (const float*)d_in[0], IMG, NB, 3, 300, 300);
    }

    auto conv = [&](const unsigned short* x, int H, int W, int idx,
                    int stride, int pad, int dil,
                    unsigned short* ybf, float* yf, int relu,
                    int imgstride, int regoff) {
        const Spec& s = S[idx];
        int Ho = (H + 2*pad - dil*(s.k - 1) - 1) / stride + 1;
        int Wo = (W + 2*pad - dil*(s.k - 1) - 1) / stride + 1;
        long M = (long)NB * Ho * Wo;
        dim3 grid((unsigned)((M + 63) / 64), (unsigned)((s.co + 63) / 64), 1);
        conv_wmma_kernel<<<grid, 128, 0, stream>>>(
            x, (const unsigned short*)(ws + o_w[idx]), (const float*)d_in[2 + 2*idx],
            ybf, yf, NB, H, W, s.ci, s.co, Ho, Wo, s.k, stride, pad, dil,
            relu, imgstride, regoff);
    };
    auto pool = [&](const unsigned short* x, unsigned short* y, int C,
                    int H, int W, int Ho, int Wo, int k, int s, int padlo) {
        long tot = (long)NB * Ho * Wo * C;
        maxpool_kernel<<<(unsigned)((tot + 255) / 256), 256, 0, stream>>>(
            x, y, NB, C, H, W, Ho, Wo, k, s, padlo);
    };

    // ----------------------------- backbone --------------------------------
    conv(IMG, 300, 300, 0, 1, 1, 1, A,  nullptr, 1, 0, 0);   // conv1_1
    conv(A,   300, 300, 1, 1, 1, 1, Bq, nullptr, 1, 0, 0);   // conv1_2
    pool(Bq, A, 64, 300, 300, 150, 150, 2, 2, 0);
    conv(A,   150, 150, 2, 1, 1, 1, Bq, nullptr, 1, 0, 0);   // conv2_1
    conv(Bq,  150, 150, 3, 1, 1, 1, A,  nullptr, 1, 0, 0);   // conv2_2
    pool(A, Bq, 128, 150, 150, 75, 75, 2, 2, 0);
    conv(Bq,  75, 75, 4, 1, 1, 1, A,  nullptr, 1, 0, 0);     // conv3_1
    conv(A,   75, 75, 5, 1, 1, 1, Bq, nullptr, 1, 0, 0);     // conv3_2
    conv(Bq,  75, 75, 6, 1, 1, 1, A,  nullptr, 1, 0, 0);     // conv3_3
    pool(A, Bq, 256, 75, 75, 38, 38, 2, 2, 0);               // ceil-mode
    conv(Bq,  38, 38, 7, 1, 1, 1, A,  nullptr, 1, 0, 0);     // conv4_1
    conv(A,   38, 38, 8, 1, 1, 1, Bq, nullptr, 1, 0, 0);     // conv4_2
    conv(Bq,  38, 38, 9, 1, 1, 1, F4, nullptr, 1, 0, 0);     // conv4_3 -> f4
    pool(F4, A, 512, 38, 38, 19, 19, 2, 2, 0);
    conv(A,   19, 19, 10, 1, 1, 1, Bq, nullptr, 1, 0, 0);    // conv5_1
    conv(Bq,  19, 19, 11, 1, 1, 1, A,  nullptr, 1, 0, 0);    // conv5_2
    conv(A,   19, 19, 12, 1, 1, 1, Bq, nullptr, 1, 0, 0);    // conv5_3
    pool(Bq, A, 512, 19, 19, 19, 19, 3, 1, 1);
    conv(A,   19, 19, 13, 1, 6, 6, Bq, nullptr, 1, 0, 0);    // conv6 (dil 6)
    conv(Bq,  19, 19, 14, 1, 0, 1, F7, nullptr, 1, 0, 0);    // conv7 -> f7
    conv(F7,  19, 19, 15, 1, 0, 1, A,  nullptr, 1, 0, 0);    // conv8_1
    conv(A,   19, 19, 16, 2, 1, 1, F8, nullptr, 1, 0, 0);    // conv8_2 -> f8
    conv(F8,  10, 10, 17, 1, 0, 1, A,  nullptr, 1, 0, 0);    // conv9_1
    conv(A,   10, 10, 18, 2, 1, 1, F9, nullptr, 1, 0, 0);    // conv9_2 -> f9
    conv(F9,   5,  5, 19, 1, 0, 1, A,  nullptr, 1, 0, 0);    // conv10_1
    conv(A,    5,  5, 20, 1, 0, 1, F10, nullptr, 1, 0, 0);   // conv10_2 -> f10
    conv(F10,  3,  3, 21, 1, 0, 1, A,  nullptr, 1, 0, 0);    // conv11_1
    conv(A,    3,  3, 22, 1, 0, 1, F11, nullptr, 1, 0, 0);   // conv11_2 -> f11

    // ------------------------------ heads -----------------------------------
    float* locs = (float*)d_out;                         // (8, 8732, 4)
    float* cls  = (float*)d_out + (size_t)8 * 8732 * 4;  // (8, 8732, 21)
    const int LS = 8732 * 4, CS = 8732 * 21;
    conv(F4,  38, 38, 23, 1, 1, 1, nullptr, locs, 0, LS, 0);
    conv(F7,  19, 19, 24, 1, 1, 1, nullptr, locs, 0, LS, 5776 * 4);
    conv(F8,  10, 10, 25, 1, 1, 1, nullptr, locs, 0, LS, 7942 * 4);
    conv(F9,   5,  5, 26, 1, 1, 1, nullptr, locs, 0, LS, 8542 * 4);
    conv(F10,  3,  3, 27, 1, 1, 1, nullptr, locs, 0, LS, 8692 * 4);
    conv(F11,  1,  1, 28, 1, 1, 1, nullptr, locs, 0, LS, 8728 * 4);
    conv(F4,  38, 38, 29, 1, 1, 1, nullptr, cls, 0, CS, 0);
    conv(F7,  19, 19, 30, 1, 1, 1, nullptr, cls, 0, CS, 5776 * 21);
    conv(F8,  10, 10, 31, 1, 1, 1, nullptr, cls, 0, CS, 7942 * 21);
    conv(F9,   5,  5, 32, 1, 1, 1, nullptr, cls, 0, CS, 8542 * 21);
    conv(F10,  3,  3, 33, 1, 1, 1, nullptr, cls, 0, CS, 8692 * 21);
    conv(F11,  1,  1, 34, 1, 1, 1, nullptr, cls, 0, CS, 8728 * 21);
}
